// TransformerBlock_15427522527869
// MI455X (gfx1250) — matrix operand order repaired and checked by reading, not compiled, not run on hardware
//
#include <hip/hip_runtime.h>
#include <hip/hip_bf16.h>

// ---------------------------------------------------------------------------
// MI455X (gfx1250, wave32) implementation of the TransformerBlock reference.
// All GEMM-like work (QKVV proj, EPA attention matmuls, 3x3x3 convs as
// implicit GEMM, 1x1x1 conv) runs on v_wmma_f32_16x16x32_bf16.
// ---------------------------------------------------------------------------

typedef __attribute__((ext_vector_type(16))) __bf16 bf16x16;
typedef __attribute__((ext_vector_type(8)))  float  f32x8;
typedef __attribute__((ext_vector_type(8)))  unsigned short us8;

union FragU { unsigned short us[16]; bf16x16 v; };

__device__ inline unsigned short f2b(float f) {
  unsigned u = __float_as_uint(f);
  unsigned r = u + 0x7FFFu + ((u >> 16) & 1u);   // round-to-nearest-even
  return (unsigned short)(r >> 16);
}
__device__ inline float b2f(unsigned short h) {
  return __uint_as_float(((unsigned)h) << 16);
}

// A fragment: 16x32 bf16 tile, row-major in LDS with row stride lda (shorts).
// wave32 layout: lanes 0-15 -> M=lane, K=0..15; lanes 16-31 -> K=16..31.
__device__ inline bf16x16 frag_a(const unsigned short* A, int lda, int lane) {
  int m  = lane & 15;
  int kb = (lane < 16) ? 0 : 16;
  const unsigned short* p = A + m * lda + kb;
  FragU f;
#pragma unroll
  for (int i = 0; i < 16; ++i) f.us[i] = p[i];
  return f.v;
}

// B fragment: 32x16 bf16 tile staged N-major in LDS (Bt[n][k], stride ldb).
// lanes 0-15 -> N=lane, K=0..15; lanes 16-31 -> K=16..31.
__device__ inline bf16x16 frag_b(const unsigned short* Bt, int ldb, int lane) {
  int n  = lane & 15;
  int ko = (lane < 16) ? 0 : 16;
  const unsigned short* p = Bt + n * ldb + ko;
  FragU f;
#pragma unroll
  for (int i = 0; i < 16; ++i) f.us[i] = p[i];
  return f.v;
}

__device__ inline f32x8 wmma_bf16(bf16x16 a, bf16x16 b, f32x8 c) {
  return __builtin_amdgcn_wmma_f32_16x16x32_bf16(false, a, false, b, (short)0, c, false, false);
}

// D-tile (16x16 f32) lane mapping: elem i -> row mb+i, col = lane&15.
#define D_MB(lane)  (((lane) < 16) ? 0 : 8)
#define D_NL(lane)  ((lane) & 15)

// Problem constants
#define BB 8
#define CC 256
#define NN 4096
#define NH 8
#define DD 32
#define PP 64
#define BCN 1048576   // CC*NN

// ---------------------------------------------------------------------------
// 1) LayerNorm over C, transpose [B,C,N] -> bf16 [B,N,C]
// ---------------------------------------------------------------------------
__global__ void k_layernorm(const float* __restrict__ x, const float* __restrict__ lnw,
                            const float* __restrict__ lnb, unsigned short* __restrict__ out) {
  __shared__ float tile[256][17];
  int blk = blockIdx.x;                 // 2048 = B * (N/16)
  int b = blk >> 8;
  int n0 = (blk & 255) << 4;
  int t = threadIdx.x;
  const float* xb = x + (size_t)b * BCN;
#pragma unroll
  for (int j = 0; j < 16; ++j) tile[t][j] = xb[(size_t)t * NN + n0 + j];
  __syncthreads();
  int nl = t >> 4, cp = t & 15;
  float s = 0.f, sq = 0.f;
#pragma unroll
  for (int i = 0; i < 16; ++i) { float v = tile[cp + 16 * i][nl]; s += v; sq += v * v; }
#pragma unroll
  for (int m = 8; m >= 1; m >>= 1) { s += __shfl_xor(s, m, 16); sq += __shfl_xor(sq, m, 16); }
  float mean = s * (1.f / 256.f);
  float var  = sq * (1.f / 256.f) - mean * mean;
  float rs = rsqrtf(var + 1e-5f);
  unsigned short* orow = out + (size_t)(b * NN + n0 + nl) * CC;
#pragma unroll
  for (int i = 0; i < 16; ++i) {
    int c = cp + 16 * i;
    orow[c] = f2b((tile[c][nl] - mean) * rs * lnw[c] + lnb[c]);
  }
}

// ---------------------------------------------------------------------------
// 2) QKVV GEMM: [32768,256] bf16 @ [256,1024] f32(->bf16) -> bf16 [32768,1024]
//    block tile 128x64, 8 waves (4x2), each wave 32x32 (2x2 WMMA tiles)
// ---------------------------------------------------------------------------
__global__ void k_gemm_qkvv(const unsigned short* __restrict__ A, const float* __restrict__ W,
                            unsigned short* __restrict__ Dst) {
  __shared__ __attribute__((aligned(32))) unsigned short As[128 * 40];
  __shared__ __attribute__((aligned(32))) unsigned short Bs[64 * 48];
  int m0 = blockIdx.x * 128;
  int n0 = blockIdx.y * 64;
  int t = threadIdx.x, lane = t & 31, w = t >> 5;
  int wm = w & 3, wn = w >> 2;
  f32x8 acc[2][2] = {};
  for (int k0 = 0; k0 < 256; k0 += 32) {
    {
      int row = t >> 1, coff = (t & 1) * 16;
      const us8* g = (const us8*)(A + (size_t)(m0 + row) * 256 + k0 + coff);
      us8* sp = (us8*)&As[row * 40 + coff];
      sp[0] = g[0]; sp[1] = g[1];
      if (k0 + 32 < 256)
        __builtin_prefetch(A + (size_t)(m0 + row) * 256 + k0 + 32 + coff, 0, 1);
    }
#pragma unroll
    for (int i = 0; i < 8; ++i) {
      int idx = t + i * 256;
      int nn = idx & 63, kk = idx >> 6;
      Bs[nn * 48 + kk] = f2b(W[(size_t)(k0 + kk) * 1024 + n0 + nn]);
    }
    __syncthreads();
    bf16x16 a0 = frag_a(&As[(wm * 32) * 40], 40, lane);
    bf16x16 a1 = frag_a(&As[(wm * 32 + 16) * 40], 40, lane);
    bf16x16 b0 = frag_b(&Bs[(wn * 32) * 48], 48, lane);
    bf16x16 b1 = frag_b(&Bs[(wn * 32 + 16) * 48], 48, lane);
    acc[0][0] = wmma_bf16(a0, b0, acc[0][0]);
    acc[0][1] = wmma_bf16(a0, b1, acc[0][1]);
    acc[1][0] = wmma_bf16(a1, b0, acc[1][0]);
    acc[1][1] = wmma_bf16(a1, b1, acc[1][1]);
    __syncthreads();
  }
  int mb = D_MB(lane), nl = D_NL(lane);
#pragma unroll
  for (int im = 0; im < 2; ++im)
#pragma unroll
    for (int in_ = 0; in_ < 2; ++in_)
#pragma unroll
      for (int i = 0; i < 8; ++i) {
        int m = m0 + wm * 32 + im * 16 + mb + i;
        int c = n0 + wn * 32 + in_ * 16 + nl;
        Dst[(size_t)m * 1024 + c] = f2b(acc[im][in_][i]);
      }
}

// ---------------------------------------------------------------------------
// 3) L2 norms of q and k rows over N (per (b,h,dd)); stores 1/max(norm,eps)
// ---------------------------------------------------------------------------
__global__ void k_rownorm(const unsigned short* __restrict__ qkvv, float* __restrict__ rinv) {
  __shared__ float red[256];
  int blk = blockIdx.x;                 // 4096: qk*2048 + b*256 + h*32 + dd
  int qk = blk >> 11;
  int rest = blk & 2047;
  int b = rest >> 8, hd = rest & 255;
  int c = qk * 256 + hd;
  int t = threadIdx.x;
  const unsigned short* base = qkvv + (size_t)b * NN * 1024 + c;
  float s = 0.f;
  for (int i = 0; i < 16; ++i) {
    float v = b2f(base[(size_t)(t + i * 256) * 1024]);
    s += v * v;
  }
  red[t] = s; __syncthreads();
  for (int st = 128; st >= 1; st >>= 1) {
    if (t < st) red[t] += red[t + st];
    __syncthreads();
  }
  if (t == 0) rinv[blk] = 1.f / fmaxf(sqrtf(red[0]), 1e-12f);
}

// ---------------------------------------------------------------------------
// 4) EF projections: k_proj / v_proj = (32 x 4096) @ (4096 x 64) per (b,h)
// ---------------------------------------------------------------------------
__global__ void k_proj_ef(const unsigned short* __restrict__ qkvv, const float* __restrict__ EF,
                          float* __restrict__ kproj, float* __restrict__ vproj) {
  __shared__ __attribute__((aligned(32))) unsigned short Ak[32 * 40], Av[32 * 40], Bt[64 * 48];
  int bh = blockIdx.x;
  int b = bh >> 3, h = bh & 7;
  int t = threadIdx.x, lane = t & 31, w = t >> 5;
  int tm = w & 1, tn = w >> 1;
  f32x8 ak = {}, av = {};
  const unsigned short* qb = qkvv + (size_t)b * NN * 1024;
  for (int n0 = 0; n0 < NN; n0 += 32) {
#pragma unroll
    for (int i = 0; i < 4; ++i) {
      int idx = t + i * 256;
      int dd = idx & 31, nn = idx >> 5;
      size_t g = (size_t)(n0 + nn) * 1024 + h * 32 + dd;
      Ak[dd * 40 + nn] = qb[g + 256];   // k (unnormalized, per original order)
      Av[dd * 40 + nn] = qb[g + 768];   // v_sa
    }
#pragma unroll
    for (int i = 0; i < 8; ++i) {
      int idx = t + i * 256;
      int p = idx & 63, nn = idx >> 6;
      Bt[p * 48 + nn] = f2b(EF[(size_t)(n0 + nn) * 64 + p]);
    }
    __syncthreads();
    bf16x16 a1 = frag_a(&Ak[tm * 16 * 40], 40, lane);
    bf16x16 a2 = frag_a(&Av[tm * 16 * 40], 40, lane);
    bf16x16 bb = frag_b(&Bt[tn * 16 * 48], 48, lane);
    ak = wmma_bf16(a1, bb, ak);
    av = wmma_bf16(a2, bb, av);
    __syncthreads();
  }
  int mb = D_MB(lane), nl = D_NL(lane);
  float* kp = kproj + (size_t)bh * DD * PP;
  float* vp = vproj + (size_t)bh * DD * PP;
#pragma unroll
  for (int i = 0; i < 8; ++i) {
    int dd = tm * 16 + mb + i, p = tn * 16 + nl;
    kp[dd * PP + p] = ak[i];
    vp[dd * PP + p] = av[i];
  }
}

// ---------------------------------------------------------------------------
// 5) Channel attention per (b,h): S=qn@kn^T (K=4096), softmax, x_ca=P@v_ca
//    writes epa[b,n,h*32+dd]  (full coverage, plain stores)
// ---------------------------------------------------------------------------
__global__ void k_ca(const unsigned short* __restrict__ qkvv, const float* __restrict__ rinv,
                     const float* __restrict__ temp, float* __restrict__ epa) {
  __shared__ __attribute__((aligned(32))) unsigned short Aq[32 * 40], Bk[32 * 48];
  __shared__ __attribute__((aligned(32))) unsigned short Pb[32 * 40], Bv[64 * 48];
  __shared__ float Sf[32][33];
  __shared__ float qi[32], ki[32];
  int bh = blockIdx.x;
  int b = bh >> 3, h = bh & 7;
  int t = threadIdx.x, lane = t & 31, w = t >> 5;
  if (t < 32) {
    qi[t] = rinv[b * 256 + h * 32 + t];
    ki[t] = rinv[2048 + b * 256 + h * 32 + t];
  }
  __syncthreads();
  const unsigned short* qb = qkvv + (size_t)b * NN * 1024;
  f32x8 acc = {};
  int tm = w & 1, tn = (w >> 1) & 1;    // waves 0..3 own the 32x32 S tiles
  for (int n0 = 0; n0 < NN; n0 += 32) {
#pragma unroll
    for (int i = 0; i < 4; ++i) {
      int idx = t + i * 256;
      int dd = idx & 31, nn = idx >> 5;
      size_t g = (size_t)(n0 + nn) * 1024 + h * 32 + dd;
      Aq[dd * 40 + nn] = f2b(b2f(qb[g]) * qi[dd]);
      Bk[dd * 48 + nn] = f2b(b2f(qb[g + 256]) * ki[dd]);
    }
    __syncthreads();
    if (w < 4) {
      bf16x16 a  = frag_a(&Aq[tm * 16 * 40], 40, lane);
      bf16x16 bb = frag_b(&Bk[tn * 16 * 48], 48, lane);
      acc = wmma_bf16(a, bb, acc);
    }
    __syncthreads();
  }
  int mb = D_MB(lane), nl = D_NL(lane);
  float tv = temp[h];
  if (w < 4) {
#pragma unroll
    for (int i = 0; i < 8; ++i) Sf[tm * 16 + mb + i][tn * 16 + nl] = acc[i] * tv;
  }
  __syncthreads();
  if (t < 32) {
    float mx = -1e30f;
    for (int e = 0; e < 32; ++e) mx = fmaxf(mx, Sf[t][e]);
    float sum = 0.f;
    for (int e = 0; e < 32; ++e) sum += __expf(Sf[t][e] - mx);
    float inv = 1.f / sum;
    for (int e = 0; e < 32; ++e) Pb[t * 40 + e] = f2b(__expf(Sf[t][e] - mx) * inv);
  }
  __syncthreads();
  int wm = w & 1, wn2 = w >> 1;         // 8 waves -> 32x64 output tile
  for (int n0 = 0; n0 < NN; n0 += 64) {
#pragma unroll
    for (int i = 0; i < 8; ++i) {
      int idx = t + i * 256;
      int ncol = idx & 63, e = idx >> 6;
      Bv[ncol * 48 + e] = qb[(size_t)(n0 + ncol) * 1024 + 512 + h * 32 + e];  // v_ca
    }
    __syncthreads();
    f32x8 o = {};
    bf16x16 a  = frag_a(&Pb[wm * 16 * 40], 40, lane);
    bf16x16 bb = frag_b(&Bv[wn2 * 16 * 48], 48, lane);
    o = wmma_bf16(a, bb, o);
#pragma unroll
    for (int i = 0; i < 8; ++i) {
      int dd = wm * 16 + mb + i;
      int n = n0 + wn2 * 16 + nl;
      epa[((size_t)b * NN + n) * CC + h * 32 + dd] = o[i];
    }
    __syncthreads();
  }
}

// ---------------------------------------------------------------------------
// 6) Spatial (projected) attention per (b,h,n-chunk of 64); scatter-add with
//    the reference's faithful [B,d,h,N]->[B,N,C] permutation.
// ---------------------------------------------------------------------------
__global__ void k_sa(const unsigned short* __restrict__ qkvv, const float* __restrict__ rinv,
                     const float* __restrict__ kproj, const float* __restrict__ vproj,
                     const float* __restrict__ temp2, float* __restrict__ epa) {
  __shared__ __attribute__((aligned(32))) unsigned short Aq[64 * 40], Bkp[64 * 48];
  __shared__ __attribute__((aligned(32))) unsigned short Pb[64 * 72], Bvp[32 * 80];
  __shared__ float Sf[64][65];
  __shared__ float qi[32];
  int n0 = blockIdx.x * 64;
  int h = blockIdx.y, b = blockIdx.z;
  int bh = b * 8 + h;
  int t = threadIdx.x, lane = t & 31, w = t >> 5;
  if (t < 32) qi[t] = rinv[b * 256 + h * 32 + t];
  __syncthreads();
  const unsigned short* qb = qkvv + (size_t)b * NN * 1024;
#pragma unroll
  for (int i = 0; i < 8; ++i) {          // A = qn^T : [64 n][32 dd]
    int idx = t + i * 256;
    int dd = idx & 31, nr = idx >> 5;
    Aq[nr * 40 + dd] = f2b(b2f(qb[(size_t)(n0 + nr) * 1024 + h * 32 + dd]) * qi[dd]);
  }
#pragma unroll
  for (int i = 0; i < 8; ++i) {          // Bt[p][dd] <- k_proj
    int idx = t + i * 256;
    int p = idx & 63, dd = idx >> 6;
    Bkp[p * 48 + dd] = f2b(kproj[(size_t)bh * 2048 + dd * 64 + p]);
  }
  __syncthreads();
  int wm = w & 3, wn = w >> 2;
  float t2 = temp2[h];
  {
    bf16x16 a = frag_a(&Aq[wm * 16 * 40], 40, lane);
    int mb = D_MB(lane), nl = D_NL(lane);
#pragma unroll
    for (int in_ = 0; in_ < 2; ++in_) {
      f32x8 o = {};
      bf16x16 bb = frag_b(&Bkp[(wn * 32 + in_ * 16) * 48], 48, lane);
      o = wmma_bf16(a, bb, o);
#pragma unroll
      for (int i = 0; i < 8; ++i)
        Sf[wm * 16 + mb + i][wn * 32 + in_ * 16 + nl] = o[i] * t2;
    }
  }
#pragma unroll
  for (int i = 0; i < 8; ++i) {          // Bt[dd][p] <- v_proj (for P @ v_proj^T)
    int idx = t + i * 256;
    int p = idx & 63, dd = idx >> 6;
    Bvp[dd * 80 + p] = f2b(vproj[(size_t)bh * 2048 + dd * 64 + p]);
  }
  __syncthreads();
  if (t < 64) {
    float mx = -1e30f;
    for (int p = 0; p < 64; ++p) mx = fmaxf(mx, Sf[t][p]);
    float sum = 0.f;
    for (int p = 0; p < 64; ++p) sum += __expf(Sf[t][p] - mx);
    float inv = 1.f / sum;
    for (int p = 0; p < 64; ++p) Pb[t * 72 + p] = f2b(__expf(Sf[t][p] - mx) * inv);
  }
  __syncthreads();
  int wm2 = w & 3, wn2 = w >> 2;
  f32x8 o = {};
#pragma unroll
  for (int kk = 0; kk < 64; kk += 32) {
    bf16x16 a  = frag_a(&Pb[wm2 * 16 * 72 + kk], 72, lane);
    bf16x16 bb = frag_b(&Bvp[wn2 * 16 * 80 + kk], 80, lane);
    o = wmma_bf16(a, bb, o);
  }
  int mb = D_MB(lane), nl = D_NL(lane);
  float* eb = epa + (size_t)b * BCN;
#pragma unroll
  for (int i = 0; i < 8; ++i) {
    int nloc = wm2 * 16 + mb + i;
    int dd = wn2 * 16 + nl;
    size_t f = (size_t)dd * 32768 + (size_t)h * 4096 + (n0 + nloc);  // [d,h,N] flat
    eb[f] += o[i];                        // disjoint across (b,h,dd,n)
  }
}

// ---------------------------------------------------------------------------
// 7) attn residual: skip[b,c,n] = x + gamma[c]*epa[b,n,c] ; also bf16 copy
// ---------------------------------------------------------------------------
__global__ void k_skip(const float* __restrict__ x, const float* __restrict__ gamma,
                       const float* __restrict__ epa, float* __restrict__ skip,
                       unsigned short* __restrict__ skipb) {
  __shared__ float tile[64][65];
  int n0 = blockIdx.x * 64;
  int c0 = blockIdx.y * 64;
  int b = blockIdx.z;
  int t = threadIdx.x;
  const float* eb = epa + (size_t)b * BCN;
#pragma unroll
  for (int i = 0; i < 16; ++i) {
    int idx = t + i * 256;
    int cl = idx & 63, nl = idx >> 6;
    tile[cl][nl] = eb[(size_t)(n0 + nl) * CC + c0 + cl];
  }
  __syncthreads();
#pragma unroll
  for (int i = 0; i < 16; ++i) {
    int idx = t + i * 256;
    int nl = idx & 63, cl = idx >> 6;
    int c = c0 + cl;
    size_t a = (size_t)b * BCN + (size_t)c * NN + n0 + nl;
    float v = x[a] + gamma[c] * tile[cl][nl];
    skip[a] = v;
    skipb[a] = f2b(v);
  }
}

// ---------------------------------------------------------------------------
// 8) 3x3x3 conv as implicit GEMM: M=B*N spatial, Ncol=co, K=27*256.
//    Block tile 128x64, 8 waves (4x2), each wave a 32x32 accumulator
//    (4 WMMAs per K-step). Each thread owns one fixed spatial row, so the
//    tap-shifted address + pad mask is computed once per tap (hoisted out of
//    the ci loop).
// ---------------------------------------------------------------------------
__global__ void k_conv(const unsigned short* __restrict__ in, const float* __restrict__ wgt,
                       float* __restrict__ out) {
  __shared__ __attribute__((aligned(32))) unsigned short As[128 * 40];
  __shared__ __attribute__((aligned(32))) unsigned short Bs[64 * 48];
  int s0 = blockIdx.x * 128;
  int co0 = blockIdx.y * 64;
  int t = threadIdx.x, lane = t & 31, w = t >> 5;
  int wm = w & 3, wn = w >> 2;
  f32x8 acc[2][2] = {};
  int b = s0 >> 12;
  int r0 = s0 & 4095;
  int myrow = t & 127;                   // spatial row owned by this thread
  int cbase = t >> 7;                    // 0 or 1: ci parity this thread fills
  int r = r0 + myrow;
  int z0 = r >> 8, y0 = (r >> 4) & 15, x0 = r & 15;
  const unsigned short* inb = in + (size_t)b * BCN;
  for (int tap = 0; tap < 27; ++tap) {
    int dz = tap / 9 - 1, dy = (tap / 3) % 3 - 1, dx = tap % 3 - 1;
    int z = z0 + dz, y = y0 + dy, xx = x0 + dx;
    bool valid = ((unsigned)z < 16u) & ((unsigned)y < 16u) & ((unsigned)xx < 16u);
    int sp = z * 256 + y * 16 + xx;      // tap-shifted spatial offset
    for (int ci0 = 0; ci0 < 256; ci0 += 32) {
      // A tile: 128 spatial rows x 32 ci (zero-padded gather, mask hoisted)
#pragma unroll
      for (int i = 0; i < 16; ++i) {
        int cc = cbase + 2 * i;
        unsigned short v = 0;
        if (valid) v = inb[(size_t)(ci0 + cc) * NN + sp];
        As[myrow * 40 + cc] = v;
      }
      // B tile: 64 co x 32 ci (weights f32 -> bf16, staged N-major)
#pragma unroll
      for (int i = 0; i < 8; ++i) {
        int idx = t + i * 256;
        int co = idx & 63, cc = idx >> 6;
        Bs[co * 48 + cc] = f2b(wgt[(size_t)(co0 + co) * 6912 + (ci0 + cc) * 27 + tap]);
      }
      if (ci0 + 32 < 256)
        __builtin_prefetch(wgt + (size_t)(co0 + (t & 63)) * 6912 + (ci0 + 32) * 27 + tap, 0, 1);
      __syncthreads();
      bf16x16 a0 = frag_a(&As[(wm * 32) * 40], 40, lane);
      bf16x16 a1 = frag_a(&As[(wm * 32 + 16) * 40], 40, lane);
      bf16x16 b0 = frag_b(&Bs[(wn * 32) * 48], 48, lane);
      bf16x16 b1 = frag_b(&Bs[(wn * 32 + 16) * 48], 48, lane);
      acc[0][0] = wmma_bf16(a0, b0, acc[0][0]);
      acc[0][1] = wmma_bf16(a0, b1, acc[0][1]);
      acc[1][0] = wmma_bf16(a1, b0, acc[1][0]);
      acc[1][1] = wmma_bf16(a1, b1, acc[1][1]);
      __syncthreads();
    }
  }
  int mb = D_MB(lane), nl = D_NL(lane);
#pragma unroll
  for (int im = 0; im < 2; ++im)
#pragma unroll
    for (int in_ = 0; in_ < 2; ++in_)
#pragma unroll
      for (int i = 0; i < 8; ++i) {
        int s = s0 + wm * 32 + im * 16 + mb + i;
        int n = s & 4095;
        int co = co0 + wn * 32 + in_ * 16 + nl;
        out[(size_t)b * BCN + (size_t)co * NN + n] = acc[im][in_][i];
      }
}

// ---------------------------------------------------------------------------
// 9) BatchNorm batch statistics (per channel) -> scale/shift
// ---------------------------------------------------------------------------
__global__ void k_bnstats(const float* __restrict__ v, const float* __restrict__ bw,
                          const float* __restrict__ bb, float* __restrict__ bnp) {
  __shared__ float rs_[256], rq[256];
  int c = blockIdx.x, t = threadIdx.x;
  float s = 0.f, sq = 0.f;
  for (int b = 0; b < BB; ++b) {
    const float* base = v + (size_t)b * BCN + (size_t)c * NN;
    for (int i = 0; i < 16; ++i) {
      float x = base[t + i * 256];
      s += x; sq += x * x;
    }
  }
  rs_[t] = s; rq[t] = sq; __syncthreads();
  for (int st = 128; st >= 1; st >>= 1) {
    if (t < st) { rs_[t] += rs_[t + st]; rq[t] += rq[t + st]; }
    __syncthreads();
  }
  if (t == 0) {
    float mean = rs_[0] * (1.f / 32768.f);
    float var  = rq[0] * (1.f / 32768.f) - mean * mean;
    float inv = rsqrtf(var + 1e-5f);
    float sc = bw[c] * inv;
    bnp[c] = sc;
    bnp[256 + c] = bb[c] - mean * sc;
  }
}

// 10) BN + leaky-relu -> bf16 activation for next conv
__global__ void k_bn_act(const float* __restrict__ v, const float* __restrict__ bnp,
                         unsigned short* __restrict__ yb) {
  int t0 = blockIdx.x * 2048 + threadIdx.x;
#pragma unroll
  for (int i = 0; i < 8; ++i) {
    int idx = t0 + i * 256;
    int c = (idx >> 12) & 255;
    float x = v[idx] * bnp[c] + bnp[256 + c];
    yb[idx] = f2b(x > 0.f ? x : 0.01f * x);
  }
}

// 11) BN + residual add + leaky-relu -> bf16 (input to conv8)
__global__ void k_bn_add_act(const float* __restrict__ v, const float* __restrict__ bnp,
                             const float* __restrict__ skip, unsigned short* __restrict__ yb) {
  int t0 = blockIdx.x * 2048 + threadIdx.x;
#pragma unroll
  for (int i = 0; i < 8; ++i) {
    int idx = t0 + i * 256;
    int c = (idx >> 12) & 255;
    float x = v[idx] * bnp[c] + bnp[256 + c] + skip[idx];
    yb[idx] = f2b(x > 0.f ? x : 0.01f * x);
  }
}

// ---------------------------------------------------------------------------
// 12) conv8 (1x1x1) as GEMM [co,ci]@[ci,n] + bias + final residual
// ---------------------------------------------------------------------------
__global__ void k_conv8(const unsigned short* __restrict__ yb, const float* __restrict__ w8,
                        const float* __restrict__ b8, const float* __restrict__ skip,
                        float* __restrict__ out) {
  __shared__ __attribute__((aligned(32))) unsigned short As[64 * 40], Bs[64 * 48];
  int n0 = blockIdx.x * 64;
  int co0 = blockIdx.y * 64;
  int b = blockIdx.z;
  int t = threadIdx.x, lane = t & 31, w = t >> 5;
  int wm = w & 1, wn = w >> 1;
  f32x8 acc[2] = {};
  for (int ci0 = 0; ci0 < 256; ci0 += 32) {
#pragma unroll
    for (int i = 0; i < 8; ++i) {
      int idx = t + i * 256;
      int co = idx & 63, cc = idx >> 6;
      As[co * 40 + cc] = f2b(w8[(size_t)(co0 + co) * 256 + ci0 + cc]);
    }
#pragma unroll
    for (int i = 0; i < 8; ++i) {
      int idx = t + i * 256;
      int nl = idx & 63, cc = idx >> 6;
      Bs[nl * 48 + cc] = yb[(size_t)b * BCN + (size_t)(ci0 + cc) * NN + n0 + nl];
    }
    __syncthreads();
    bf16x16 bfrag = frag_b(&Bs[wn * 16 * 48], 48, lane);
#pragma unroll
    for (int im = 0; im < 2; ++im) {
      bf16x16 a = frag_a(&As[(wm * 32 + im * 16) * 40], 40, lane);
      acc[im] = wmma_bf16(a, bfrag, acc[im]);
    }
    __syncthreads();
  }
  int mb = D_MB(lane), nl = D_NL(lane);
#pragma unroll
  for (int im = 0; im < 2; ++im)
#pragma unroll
    for (int i = 0; i < 8; ++i) {
      int co = co0 + wm * 32 + im * 16 + mb + i;
      int n = n0 + wn * 16 + nl;
      size_t a = (size_t)b * BCN + (size_t)co * NN + n;
      out[a] = skip[a] + acc[im][i] + b8[co];
    }
}

// ---------------------------------------------------------------------------
extern "C" void kernel_launch(void* const* d_in, const int* in_sizes, int n_in,
                              void* d_out, int out_size, void* d_ws, size_t ws_size,
                              hipStream_t stream) {
  const float* x     = (const float*)d_in[0];
  const float* Wq    = (const float*)d_in[1];
  const float* EF    = (const float*)d_in[2];
  const float* temp  = (const float*)d_in[3];
  const float* temp2 = (const float*)d_in[4];
  const float* lnw   = (const float*)d_in[5];
  const float* lnb   = (const float*)d_in[6];
  const float* gamma = (const float*)d_in[7];
  const float* w1    = (const float*)d_in[8];
  const float* bn1w  = (const float*)d_in[9];
  const float* bn1b  = (const float*)d_in[10];
  const float* w2    = (const float*)d_in[11];
  const float* bn2w  = (const float*)d_in[12];
  const float* bn2b  = (const float*)d_in[13];
  const float* w8    = (const float*)d_in[14];
  const float* b8    = (const float*)d_in[15];
  float* out = (float*)d_out;
  (void)in_sizes; (void)n_in; (void)out_size; (void)ws_size;

  char* ws = (char*)d_ws;
  size_t off = 0;
  auto take = [&](size_t bytes) -> void* {
    void* p = ws + off;
    off += (bytes + 255) & ~(size_t)255;
    return p;
  };
  unsigned short* lnbf  = (unsigned short*)take((size_t)BB * NN * CC * 2);        // 16.8 MB
  unsigned short* qkvvb = (unsigned short*)take((size_t)BB * NN * 1024 * 2);      // 67 MB
  float* rinv  = (float*)take(4096 * 4);
  float* kproj = (float*)take((size_t)BB * NH * DD * PP * 4);
  float* vproj = (float*)take((size_t)BB * NH * DD * PP * 4);
  float* epa   = (float*)take((size_t)BB * BCN * 4);                              // 33.5 MB
  float* skip  = (float*)take((size_t)BB * BCN * 4);                              // 33.5 MB
  unsigned short* skipb = (unsigned short*)take((size_t)BB * BCN * 2);            // 16.8 MB
  float* cbuf  = (float*)take((size_t)BB * BCN * 4);                              // 33.5 MB (conv1 then conv2 out)
  unsigned short* ybf = (unsigned short*)take((size_t)BB * BCN * 2);              // 16.8 MB
  float* bnp1 = (float*)take(512 * 4);
  float* bnp2 = (float*)take(512 * 4);

  k_layernorm<<<2048, 256, 0, stream>>>(x, lnw, lnb, lnbf);
  k_gemm_qkvv<<<dim3(256, 16), 256, 0, stream>>>(lnbf, Wq, qkvvb);
  k_rownorm<<<4096, 256, 0, stream>>>(qkvvb, rinv);
  k_proj_ef<<<64, 256, 0, stream>>>(qkvvb, EF, kproj, vproj);
  k_ca<<<64, 256, 0, stream>>>(qkvvb, rinv, temp, epa);
  k_sa<<<dim3(64, 8, 8), 256, 0, stream>>>(qkvvb, rinv, kproj, vproj, temp2, epa);
  k_skip<<<dim3(64, 4, 8), 256, 0, stream>>>(x, gamma, epa, skip, skipb);
  k_conv<<<dim3(256, 4), 256, 0, stream>>>(skipb, w1, cbuf);
  k_bnstats<<<256, 256, 0, stream>>>(cbuf, bn1w, bn1b, bnp1);
  k_bn_act<<<4096, 256, 0, stream>>>(cbuf, bnp1, ybf);
  k_conv<<<dim3(256, 4), 256, 0, stream>>>(ybf, w2, cbuf);
  k_bnstats<<<256, 256, 0, stream>>>(cbuf, bn2w, bn2b, bnp2);
  k_bn_add_act<<<4096, 256, 0, stream>>>(cbuf, bnp2, skip, ybf);
  k_conv8<<<dim3(64, 4, 8), 256, 0, stream>>>(ybf, w8, b8, skip, out);
}
